// LocalAttention_14328010900303
// MI455X (gfx1250) — compile-verified, matrix operand order checked by reading, not run on hardware
//
#include <hip/hip_runtime.h>

// B=64, L=1024, E=512, WIN=5, C=200
#define Bn 64
#define Ln 1024
#define En 512
#define Cn 200
#define CP 208            // C padded to 13 n-tiles of 16
#define ROWS 64           // positions per block
#define STR 520           // LDS row stride in halves (512 + 8 pad -> conflict-free b128)

typedef __attribute__((ext_vector_type(16))) _Float16 v16h;
typedef __attribute__((ext_vector_type(8)))  float    v8f;

union F16x16 { v16h v; int4 q[2]; };
union H4 { _Float16 h[4]; int2 p; };

__device__ __forceinline__ unsigned f2ord(float f) {
    unsigned b = __float_as_uint(f);
    return (b & 0x80000000u) ? ~b : (b | 0x80000000u);
}
__device__ __forceinline__ float ord2f(unsigned u) {
    unsigned b = (u & 0x80000000u) ? (u & 0x7FFFFFFFu) : ~u;
    return __uint_as_float(b);
}

__global__ void la_init(unsigned* o, int n) {
    int i = blockIdx.x * 256 + threadIdx.x;
    if (i < n) o[i] = f2ord(-2.0f);
}
__global__ void la_fin(unsigned* o, int n) {
    int i = blockIdx.x * 256 + threadIdx.x;
    if (i < n) o[i] = __float_as_uint(ord2f(o[i]));
}

__global__ __launch_bounds__(256) void la_main(
    const float* __restrict__ x,      // [B,L,E]
    const float* __restrict__ W_att,  // [5,E]
    const float* __restrict__ b_att,  // [1]
    const float* __restrict__ W_cnn,  // [C,E]
    const float* __restrict__ b_cnn,  // [C]
    unsigned* __restrict__ outU)      // [B,C] ordered-uint encoded
{
    __shared__ _Float16 aT[ROWS * STR];        // gated x tile, f16
    __shared__ _Float16 bT[CP * STR];          // W_cnn, f16 (rows >= C zeroed)
    __shared__ float    wAtt[5 * En];
    __shared__ float    rdot[(ROWS + 4) * 5];
    __shared__ float    sc[ROWS];
    __shared__ unsigned colMax[CP];

    const int tid  = threadIdx.x;
    const int wave = tid >> 5;
    const int lane = tid & 31;
    const int b    = blockIdx.y;
    const int l0   = blockIdx.x * ROWS;
    const float* xb = x + (size_t)b * (Ln * En);

    // ---- stage W_att, init column maxima ----
    if (tid < CP) colMax[tid] = f2ord(-2.0f);
    for (int i = tid; i < 5 * En; i += 256) wAtt[i] = W_att[i];
    __syncthreads();

    // ---- per-row dots with the 5 attention filters (rows l0-2 .. l0+65) ----
    for (int r = wave; r < ROWS + 4; r += 8) {
        int j = l0 - 2 + r;
        float s0 = 0.f, s1 = 0.f, s2 = 0.f, s3 = 0.f, s4 = 0.f;
        if (j >= 0 && j < Ln) {
            const float* xr = xb + (size_t)j * En + lane * 16;
            const float* w0 = &wAtt[0 * En + lane * 16];
            const float* w1 = &wAtt[1 * En + lane * 16];
            const float* w2 = &wAtt[2 * En + lane * 16];
            const float* w3 = &wAtt[3 * En + lane * 16];
            const float* w4 = &wAtt[4 * En + lane * 16];
#pragma unroll
            for (int u = 0; u < 16; u += 4) {
                float4 v = *(const float4*)(xr + u);
                float4 a0 = *(const float4*)(w0 + u);
                float4 a1 = *(const float4*)(w1 + u);
                float4 a2 = *(const float4*)(w2 + u);
                float4 a3 = *(const float4*)(w3 + u);
                float4 a4 = *(const float4*)(w4 + u);
                s0 += v.x * a0.x + v.y * a0.y + v.z * a0.z + v.w * a0.w;
                s1 += v.x * a1.x + v.y * a1.y + v.z * a1.z + v.w * a1.w;
                s2 += v.x * a2.x + v.y * a2.y + v.z * a2.z + v.w * a2.w;
                s3 += v.x * a3.x + v.y * a3.y + v.z * a3.z + v.w * a3.w;
                s4 += v.x * a4.x + v.y * a4.y + v.z * a4.z + v.w * a4.w;
            }
#pragma unroll
            for (int off = 16; off > 0; off >>= 1) {
                s0 += __shfl_xor(s0, off);
                s1 += __shfl_xor(s1, off);
                s2 += __shfl_xor(s2, off);
                s3 += __shfl_xor(s3, off);
                s4 += __shfl_xor(s4, off);
            }
        }
        if (lane == 0) {
            rdot[r * 5 + 0] = s0; rdot[r * 5 + 1] = s1; rdot[r * 5 + 2] = s2;
            rdot[r * 5 + 3] = s3; rdot[r * 5 + 4] = s4;
        }
    }
    __syncthreads();

    // ---- scores = sigmoid(window-sum + b_att) ----
    if (tid < ROWS) {
        float s = b_att[0];
#pragma unroll
        for (int w = 0; w < 5; ++w) s += rdot[(tid + w) * 5 + w];
        sc[tid] = 1.0f / (1.0f + expf(-s));
    }
    __syncthreads();

    // ---- fill A tile (gated x -> f16) ----
    for (int i = tid; i < ROWS * (En / 4); i += 256) {
        int row = i >> 7;
        int e4  = (i & 127) * 4;
        float4 v = *(const float4*)(xb + (size_t)(l0 + row) * En + e4);
        float s = sc[row];
        H4 pk;
        pk.h[0] = (_Float16)(v.x * s); pk.h[1] = (_Float16)(v.y * s);
        pk.h[2] = (_Float16)(v.z * s); pk.h[3] = (_Float16)(v.w * s);
        *(int2*)&aT[row * STR + e4] = pk.p;
    }
    // ---- fill B tile (W_cnn -> f16, padded rows zero) ----
    for (int i = tid; i < CP * (En / 4); i += 256) {
        int row = i >> 7;
        int e4  = (i & 127) * 4;
        H4 pk;
        if (row < Cn) {
            float4 v = *(const float4*)(W_cnn + (size_t)row * En + e4);
            pk.h[0] = (_Float16)v.x; pk.h[1] = (_Float16)v.y;
            pk.h[2] = (_Float16)v.z; pk.h[3] = (_Float16)v.w;
        } else {
            pk.h[0] = pk.h[1] = pk.h[2] = pk.h[3] = (_Float16)0.0f;
        }
        *(int2*)&bT[row * STR + e4] = pk.p;
    }
    __syncthreads();

    // ---- WMMA GEMM: 4 M-tiles x 13 N-tiles, K=512 ----
    const int hi = (lane >= 16) ? 8 : 0;
    for (int t = wave; t < (ROWS / 16) * (CP / 16); t += 8) {
        int m0 = (t & 3) << 4;
        int n0 = (t >> 2) << 4;
        const _Float16* ap = &aT[(m0 + (lane & 15)) * STR + hi];
        const _Float16* bp = &bT[(n0 + (lane & 15)) * STR + hi];
        v8f acc = {};
#pragma unroll
        for (int k0 = 0; k0 < En; k0 += 32) {
            F16x16 af, bf;
            af.q[0] = *(const int4*)(ap + k0);
            af.q[1] = *(const int4*)(ap + k0 + 16);
            bf.q[0] = *(const int4*)(bp + k0);
            bf.q[1] = *(const int4*)(bp + k0 + 16);
            acc = __builtin_amdgcn_wmma_f32_16x16x32_f16(
                false, af.v, false, bf.v, (short)0, acc, false, false);
        }
        int c = n0 + (lane & 15);
        float m = acc[0];
#pragma unroll
        for (int i = 1; i < 8; ++i) m = fmaxf(m, acc[i]);
        float bias = (c < Cn) ? b_cnn[c] : 0.0f;
        float val = tanhf(m + bias);            // tanh monotone: max(tanh)=tanh(max)
        val = fmaxf(val, __shfl_xor(val, 16));  // combine M 0-7 with M 8-15
        if (lane < 16 && c < Cn) atomicMax(&colMax[c], f2ord(val));
    }
    __syncthreads();

    if (tid < Cn) atomicMax(outU + b * Cn + tid, colMax[tid]);
}

extern "C" void kernel_launch(void* const* d_in, const int* in_sizes, int n_in,
                              void* d_out, int out_size, void* d_ws, size_t ws_size,
                              hipStream_t stream) {
    const float* x     = (const float*)d_in[0];
    const float* W_att = (const float*)d_in[1];
    const float* b_att = (const float*)d_in[2];
    const float* W_cnn = (const float*)d_in[3];
    const float* b_cnn = (const float*)d_in[4];
    unsigned* outU = (unsigned*)d_out;

    const int n = Bn * Cn; // 12800
    la_init<<<(n + 255) / 256, 256, 0, stream>>>(outU, n);
    dim3 grid(Ln / ROWS, Bn); // (16, 64)
    la_main<<<grid, 256, 0, stream>>>(x, W_att, b_att, W_cnn, b_cnn, outU);
    la_fin<<<(n + 255) / 256, 256, 0, stream>>>(outU, n);
}
